// MultiHeadAttention_3770981285897
// MI455X (gfx1250) — matrix operand order repaired and checked by reading, not compiled, not run on hardware
//
#include <hip/hip_runtime.h>

#define SEQ 4096
#define DM  512
#define DK  64
#define NH  8

typedef __attribute__((ext_vector_type(16))) _Float16 v16h;
typedef __attribute__((ext_vector_type(8)))  _Float16 v8h;
typedef __attribute__((ext_vector_type(8)))  float    v8f;

// ---------------- WMMA helpers (CDNA5 v_wmma_f32_16x16x32_f16) ----------------

__device__ __forceinline__ v8f wmma16(v16h a, v16h b, v8f c) {
  return __builtin_amdgcn_wmma_f32_16x16x32_f16(
      /*neg_a=*/false, a, /*neg_b=*/false, b,
      /*c_mod=*/(short)0, c, /*reuse_a=*/false, /*reuse_b=*/false);
}

// A-matrix (16x32 f16) per-lane fragment: lane L holds row M=L&15.
// lanes 0-15: elems 0..7 = K 0..7,  elems 8..15 = K 16..23
// lanes16-31: elems 0..7 = K 8..15, elems 8..15 = K 24..31
__device__ __forceinline__ v16h loadA(const _Float16* rowp, int k0, int koff) {
  v8h lo = *(const v8h*)(rowp + k0 + koff);
  v8h hi = *(const v8h*)(rowp + k0 + koff + 16);
  return __builtin_shufflevector(lo, hi, 0,1,2,3,4,5,6,7,8,9,10,11,12,13,14,15);
}

// B-matrix (32x16 f16): lane L holds column N=L&15; half h: elems 0..7 = K 8h..8h+7, elems 8..15 = K 16+8h..23+8h.
__device__ __forceinline__ v16h loadB(const _Float16* p, int ld) {
  const int hb = (threadIdx.x >> 4) & 1;
  v16h r;
  #pragma unroll
  for (int s = 0; s < 8; s++) {
    r[s]     = p[(size_t)(8 * hb + s) * ld];
    r[8 + s] = p[(size_t)(16 + 8 * hb + s) * ld];
  }
  return r;
}

// 16-lane (half-wave) reductions — keep the two 16-lane halves independent.
__device__ __forceinline__ float hmax16(float v) {
  #pragma unroll
  for (int m = 1; m < 16; m <<= 1) v = fmaxf(v, __shfl_xor(v, m, 32));
  return v;
}
__device__ __forceinline__ float hsum16(float v) {
  #pragma unroll
  for (int m = 1; m < 16; m <<= 1) v += __shfl_xor(v, m, 32);
  return v;
}

// Online (max, sumexp) update with a SINGLE v_exp_f32 per element:
// d = s-m; e = exp(-|d|); (m,l) = d>0 ? (s, l*e+1) : (m, l+e)
__device__ __forceinline__ void online_update(float s, float& m, float& l) {
  float d  = s - m;
  float e  = __expf(-fabsf(d));
  bool  gt = d > 0.0f;
  m = gt ? s : m;
  l = gt ? __builtin_fmaf(l, e, 1.0f) : (l + e);
}

// ---------------- Kernel 0: fp32 -> f16 staging ----------------

__global__ void convert_kernel(const float* __restrict__ X,
                               const float* __restrict__ wq,
                               const float* __restrict__ wk,
                               const float* __restrict__ wv,
                               const float* __restrict__ pw,
                               _Float16* __restrict__ Xh,
                               _Float16* __restrict__ Wq,
                               _Float16* __restrict__ Wk,
                               _Float16* __restrict__ Wv,
                               _Float16* __restrict__ PWT) {
  int idx = blockIdx.x * blockDim.x + threadIdx.x;
  if (idx < SEQ * DM) Xh[idx] = (_Float16)X[idx];
  if (idx < NH * DM * DK) {
    Wq[idx] = (_Float16)wq[idx];
    Wk[idx] = (_Float16)wk[idx];
    Wv[idx] = (_Float16)wv[idx];
  }
  if (idx < DM * DM) {           // proj_w [512][512] -> transposed f16
    int kk = idx >> 9, n = idx & 511;
    PWT[idx] = (_Float16)pw[n * DM + kk];
  }
}

// ---------------- Kernel 1: QKV projection (WMMA) ----------------
// grid (SEQ/64, NH, 3); block 128 (4 waves x 16 rows).
// z=0 -> Qs [h][seq][dk], z=1 -> KsT [h][dk][seq] (transposed), z=2 -> Vs [h][seq][dk]

__global__ __launch_bounds__(128) void qkv_kernel(
    const _Float16* __restrict__ Xh, const _Float16* __restrict__ Wbase,
    _Float16* __restrict__ Qs, _Float16* __restrict__ KsT,
    _Float16* __restrict__ Vs) {
  const int lane = threadIdx.x & 31;
  const int w    = threadIdx.x >> 5;
  const int half = lane >> 4;
  const int koff = half ? 8 : 0;
  const int n16  = lane & 15;
  const int h    = blockIdx.y;
  const int z    = blockIdx.z;
  const int row0 = blockIdx.x * 64 + w * 16;

  const _Float16* W    = Wbase + (size_t)z * NH * DM * DK + (size_t)h * DM * DK;
  const _Float16* arow = Xh + (size_t)(row0 + n16) * DM;

  v8f acc[4];
  #pragma unroll
  for (int t = 0; t < 4; t++) acc[t] = v8f{};

  for (int k0 = 0; k0 < DM; k0 += 32) {
    v16h a = loadA(arow, k0, koff);
    #pragma unroll
    for (int t = 0; t < 4; t++) {
      v16h b = loadB(W + (size_t)k0 * DK + t * 16 + n16, DK);
      acc[t] = wmma16(a, b, acc[t]);
    }
  }

  if (z == 1) {  // K stored transposed: KsT[h][dk][seq]
    _Float16* kb = KsT + (size_t)h * DK * SEQ;
    #pragma unroll
    for (int t = 0; t < 4; t++)
      #pragma unroll
      for (int i = 0; i < 8; i++)
        kb[(size_t)(t * 16 + n16) * SEQ + row0 + i + 8 * half] =
            (_Float16)acc[t][i];
  } else {
    _Float16* ob = (z == 0 ? Qs : Vs) + (size_t)h * SEQ * DK;
    #pragma unroll
    for (int t = 0; t < 4; t++)
      #pragma unroll
      for (int i = 0; i < 8; i++)
        ob[(size_t)(row0 + i + 8 * half) * DK + t * 16 + n16] =
            (_Float16)acc[t][i];
  }
}

// ---------------- Kernel 2: attention (two-pass streaming softmax) ----------------
// grid (SEQ/64, NH); block 128 (4 waves x 16 query rows).
// Pass 1: per-lane online max/sum-exp (1 exp per element) over all keys.
// Pass 2: recompute scores, write normalized attn to d_out with NON-TEMPORAL
//         stores (537 MB streaming output must not evict the L2-resident K/V),
//         re-pack P through LDS into WMMA A-layout and accumulate O += P * V.

__global__ __launch_bounds__(128) void attn_kernel(
    const _Float16* __restrict__ Qs, const _Float16* __restrict__ KsT,
    const _Float16* __restrict__ Vs, float* __restrict__ attn_out,
    _Float16* __restrict__ Y) {
  __shared__ _Float16 lp[4 * 16 * 40];  // per-wave 16x32 P staging, stride 40 (16B aligned, bank-spread)

  const int lane = threadIdx.x & 31;
  const int w    = threadIdx.x >> 5;
  const int half = lane >> 4;
  const int koff = half ? 8 : 0;
  const int n16  = lane & 15;
  const int h    = blockIdx.y;
  const int row0 = blockIdx.x * 64 + w * 16;

  const _Float16* qrow  = Qs  + (size_t)h * SEQ * DK + (size_t)(row0 + n16) * DK;
  const _Float16* kbase = KsT + (size_t)h * DK * SEQ;
  const _Float16* vbase = Vs  + (size_t)h * SEQ * DK;
  _Float16*       lpw   = lp + w * 16 * 40;

  // Q fragment for this wave's 16 rows, all 64 features (2 k-steps), reused everywhere.
  v16h a0 = loadA(qrow, 0,  koff);
  v16h a1 = loadA(qrow, 32, koff);

  const float scale = 0.125f;  // 1/sqrt(64)

  // ---- pass 1: per-lane running (max, sumexp) over this lane's key columns ----
  float m_[8], l_[8];
  #pragma unroll
  for (int i = 0; i < 8; i++) { m_[i] = -3.0e38f; l_[i] = 0.0f; }

  for (int n0 = 0; n0 < SEQ; n0 += 32) {
    v16h b0 = loadB(kbase + n0 + n16, SEQ);
    v16h b1 = loadB(kbase + (size_t)32 * SEQ + n0 + n16, SEQ);
    v16h b2 = loadB(kbase + n0 + 16 + n16, SEQ);
    v16h b3 = loadB(kbase + (size_t)32 * SEQ + n0 + 16 + n16, SEQ);
    v8f ce = v8f{}, co = v8f{};
    ce = wmma16(a0, b0, ce);
    ce = wmma16(a1, b1, ce);
    co = wmma16(a0, b2, co);
    co = wmma16(a1, b3, co);
    #pragma unroll
    for (int i = 0; i < 8; i++) {
      online_update(ce[i] * scale, m_[i], l_[i]);
      online_update(co[i] * scale, m_[i], l_[i]);
    }
  }
  // combine across the 16 lanes of each half (rows 0-7 live in lanes 0-15, rows 8-15 in 16-31)
  float gm[8], ginv[8];
  #pragma unroll
  for (int i = 0; i < 8; i++) {
    float M = hmax16(m_[i]);
    float S = hsum16(l_[i] * __expf(m_[i] - M));
    gm[i]   = M;
    ginv[i] = 1.0f / S;
  }

  // ---- pass 2: recompute, normalize, write attn, accumulate O = P*V ----
  v8f o[4];
  #pragma unroll
  for (int t = 0; t < 4; t++) o[t] = v8f{};

  float* abase = attn_out + (size_t)h * SEQ * SEQ;

  for (int n0 = 0; n0 < SEQ; n0 += 32) {
    v8f ce = v8f{}, co = v8f{};
    {
      v16h b0 = loadB(kbase + n0 + n16, SEQ);
      v16h b1 = loadB(kbase + (size_t)32 * SEQ + n0 + n16, SEQ);
      ce = wmma16(a0, b0, ce);
      ce = wmma16(a1, b1, ce);
      v16h b2 = loadB(kbase + n0 + 16 + n16, SEQ);
      v16h b3 = loadB(kbase + (size_t)32 * SEQ + n0 + 16 + n16, SEQ);
      co = wmma16(a0, b2, co);
      co = wmma16(a1, b3, co);
    }
    #pragma unroll
    for (int i = 0; i < 8; i++) {
      int   r  = i + 8 * half;
      float pe = __expf(ce[i] * scale - gm[i]) * ginv[i];
      float po = __expf(co[i] * scale - gm[i]) * ginv[i];
      // attn write: non-temporal streaming stores (write-once, never re-read);
      // lanes 0-15 cover 16 consecutive columns -> 64B coalesced runs.
      __builtin_nontemporal_store(pe, &abase[(size_t)(row0 + r) * SEQ + n0 +      n16]);
      __builtin_nontemporal_store(po, &abase[(size_t)(row0 + r) * SEQ + n0 + 16 + n16]);
      // stage P (f16) for the A-side of P*V
      lpw[r * 40 +      n16] = (_Float16)pe;
      lpw[r * 40 + 16 + n16] = (_Float16)po;
    }
    // In-wave LDS store->load is in-order (DScnt); each wave uses its own region.
    v16h pa;
    {
      const _Float16* pr = lpw + n16 * 40 + koff;
      v8h lo = *(const v8h*)(pr);
      v8h hi = *(const v8h*)(pr + 16);
      pa = __builtin_shufflevector(lo, hi, 0,1,2,3,4,5,6,7,8,9,10,11,12,13,14,15);
    }
    #pragma unroll
    for (int t = 0; t < 4; t++) {
      v16h bv = loadB(vbase + (size_t)n0 * DK + t * 16 + n16, DK);
      o[t] = wmma16(pa, bv, o[t]);
    }
  }

  // O (16x64 f32) -> Y f16 [seq][512] head-major concat
  #pragma unroll
  for (int t = 0; t < 4; t++)
    #pragma unroll
    for (int i = 0; i < 8; i++) {
      int r = row0 + i + 8 * half;
      Y[(size_t)r * DM + h * DK + t * 16 + n16] = (_Float16)o[t][i];
    }
}

// ---------------- Kernel 3: output projection + residual + LayerNorm ----------------
// grid (SEQ/16); block 128 (4 waves). Wave w computes cols [w*128, w*128+128).

__global__ __launch_bounds__(128) void proj_ln_kernel(
    const _Float16* __restrict__ Y, const _Float16* __restrict__ PWT,
    const float* __restrict__ pb, const float* __restrict__ resid,
    const float* __restrict__ gamma, const float* __restrict__ beta,
    float* __restrict__ out) {
  __shared__ float ls[16 * 520];

  const int lane = threadIdx.x & 31;
  const int w    = threadIdx.x >> 5;
  const int half = lane >> 4;
  const int koff = half ? 8 : 0;
  const int n16  = lane & 15;
  const int r0   = blockIdx.x * 16;
  const int ncol = w * 128;

  const _Float16* arow = Y + (size_t)(r0 + n16) * DM;

  v8f acc[8];
  #pragma unroll
  for (int t = 0; t < 8; t++) acc[t] = v8f{};

  for (int k0 = 0; k0 < DM; k0 += 32) {
    v16h a = loadA(arow, k0, koff);
    #pragma unroll
    for (int t = 0; t < 8; t++) {
      v16h b = loadB(PWT + (size_t)k0 * DM + ncol + t * 16 + n16, DM);
      acc[t] = wmma16(a, b, acc[t]);
    }
  }

  // bias + residual -> LDS row buffer
  #pragma unroll
  for (int t = 0; t < 8; t++) {
    int n = ncol + t * 16 + n16;
    #pragma unroll
    for (int i = 0; i < 8; i++) {
      int r = i + 8 * half;
      ls[r * 520 + n] = acc[t][i] + pb[n] + resid[(size_t)(r0 + r) * DM + n];
    }
  }
  __syncthreads();

  // LayerNorm: wave w handles rows 4w..4w+3
  #pragma unroll
  for (int rr = 0; rr < 4; rr++) {
    int r = w * 4 + rr;
    float s = 0.0f, s2 = 0.0f;
    for (int j = lane; j < DM; j += 32) {
      float v = ls[r * 520 + j];
      s += v; s2 += v * v;
    }
    #pragma unroll
    for (int mm = 1; mm < 32; mm <<= 1) {
      s  += __shfl_xor(s,  mm, 32);
      s2 += __shfl_xor(s2, mm, 32);
    }
    float mu  = s * (1.0f / DM);
    float var = s2 * (1.0f / DM) - mu * mu;
    float rs  = rsqrtf(var + 1e-5f);
    for (int j = lane; j < DM; j += 32) {
      out[(size_t)(r0 + r) * DM + j] = (ls[r * 520 + j] - mu) * rs * gamma[j] + beta[j];
    }
  }
}

// ---------------- host launch ----------------

extern "C" void kernel_launch(void* const* d_in, const int* in_sizes, int n_in,
                              void* d_out, int out_size, void* d_ws, size_t ws_size,
                              hipStream_t stream) {
  const float* q   = (const float*)d_in[0];
  // d_in[1], d_in[2] (k, v) are ignored by the reference (q_s=k_s=v_s from q).
  const float* wq  = (const float*)d_in[3];
  const float* wk  = (const float*)d_in[4];
  const float* wv  = (const float*)d_in[5];
  const float* pw  = (const float*)d_in[6];
  const float* pb  = (const float*)d_in[7];
  const float* gm  = (const float*)d_in[8];
  const float* bt  = (const float*)d_in[9];

  float* out_main = (float*)d_out;                         // [4096,512]
  float* out_attn = (float*)d_out + (size_t)SEQ * DM;      // [8,4096,4096]

  // workspace layout (f16 elements)
  _Float16* wsb = (_Float16*)d_ws;
  _Float16* Xh  = wsb;                                     // 4096*512
  _Float16* Wqh = Xh  + (size_t)SEQ * DM;                  // 3 x 8*512*64 contiguous
  _Float16* PWT = Wqh + (size_t)3 * NH * DM * DK;          // 512*512
  _Float16* Qs  = PWT + (size_t)DM * DM;                   // 8*4096*64
  _Float16* KsT = Qs  + (size_t)NH * SEQ * DK;             // 8*64*4096
  _Float16* Vs  = KsT + (size_t)NH * DK * SEQ;             // 8*4096*64
  _Float16* Yb  = Vs  + (size_t)NH * SEQ * DK;             // 4096*512

  // 0) stage fp32 -> f16 (X, weights, transposed proj_w)
  convert_kernel<<<(SEQ * DM) / 256, 256, 0, stream>>>(
      q, wq, wk, wv, pw, Xh, Wqh, Wqh + (size_t)NH * DM * DK,
      Wqh + (size_t)2 * NH * DM * DK, PWT);

  // 1) QKV projection
  qkv_kernel<<<dim3(SEQ / 64, NH, 3), 128, 0, stream>>>(Xh, Wqh, Qs, KsT, Vs);

  // 2) attention (writes attn to d_out, head outputs to Y)
  attn_kernel<<<dim3(SEQ / 64, NH), 128, 0, stream>>>(Qs, KsT, Vs, out_attn, Yb);

  // 3) projection + residual + layernorm (writes out to d_out)
  proj_ln_kernel<<<SEQ / 16, 128, 0, stream>>>(Yb, PWT, pb, q, gm, bt, out_main);

  (void)in_sizes; (void)n_in; (void)out_size; (void)ws_size;
}
